// DecoderRNNTAtt_74947179315294
// MI455X (gfx1250) — compile-verified, matrix operand order checked by reading, not run on hardware
//
#include <hip/hip_runtime.h>
#include <hip/hip_bf16.h>

typedef __attribute__((ext_vector_type(16))) _Float16 v16h;
typedef __attribute__((ext_vector_type(8)))  float    v8f;

#define WMMA_F16(a, b, c) \
  __builtin_amdgcn_wmma_f32_16x16x32_f16(false, (a), false, (b), (short)0, (c), false, false)

// ---- constants from the reference ----
#define BB 4
#define TT 200
#define UU 64
#define EPROJS 512
#define DUNITS 512
#define EMB 256
#define ATTD 512
#define JOINT 512
#define ODIM 1000
#define ODIM_PAD 1024

// Load a 16x32 f16 WMMA fragment (A: row = lane&15 mapped by caller; B: row = n0+(lane&15)).
// ISA layout: halves 0..7 = cols k0+kg*8 .. +7 ; halves 8..15 = cols k0+16+kg*8 .. +7
// where kg = lane>>4. Two 16-byte loads.
__device__ __forceinline__ v16h ldfrag(const _Float16* p, int stride, int row, int k0) {
  int kg = (threadIdx.x >> 4) & 1;
  const _Float16* q = p + (size_t)row * (size_t)stride + k0 + kg * 8;
  union { v16h v; uint4 u[2]; } f;
  f.u[0] = *(const uint4*)(q);
  f.u[1] = *(const uint4*)(q + 16);
  return f.v;
}

__device__ __forceinline__ float sigm(float x) { return 1.0f / (1.0f + expf(-x)); }

// ---------------- conversion / embedding kernels ----------------
__global__ void rnnt_cvt_f16(const float* __restrict__ src, _Float16* __restrict__ dst, int n) {
  for (int i = blockIdx.x * blockDim.x + threadIdx.x; i < n; i += gridDim.x * blockDim.x)
    dst[i] = (_Float16)src[i];
}

__global__ void rnnt_cvt_pad_rows(const float* __restrict__ src, _Float16* __restrict__ dst,
                                  int rows_src, int rows_dst, int cols) {
  int n = rows_dst * cols;
  for (int i = blockIdx.x * blockDim.x + threadIdx.x; i < n; i += gridDim.x * blockDim.x) {
    int row = i / cols;
    dst[i] = (row < rows_src) ? (_Float16)src[i] : (_Float16)0.0f;
  }
}

__global__ void rnnt_embed(const float* __restrict__ embW, const int* __restrict__ ys,
                           _Float16* __restrict__ ey16) {
  int n = BB * UU * EMB;
  for (int i = blockIdx.x * blockDim.x + threadIdx.x; i < n; i += gridDim.x * blockDim.x) {
    int row = i >> 8;           // (b*U + u)
    int k = i & (EMB - 1);
    int y = ys[row];
    ey16[i] = (y == 0) ? (_Float16)0.0f : (_Float16)embW[y * EMB + k];
  }
}

// ---------------- generic WMMA GEMM: C(MxN) = act(A(MxK,f16) @ W(NxK,f16)^T + bias) ----------------
// grid: (N/16, M/16), block: 32 threads (one wave)
__global__ void rnnt_gemm_wmma(const _Float16* __restrict__ A, const _Float16* __restrict__ W,
                               const float* __restrict__ bias, float* __restrict__ C,
                               int K, int N, int do_tanh) {
  int lane = threadIdx.x & 31;
  int m0 = blockIdx.y * 16, n0 = blockIdx.x * 16;
  int arow = m0 + (lane & 15);
  int brow = n0 + (lane & 15);
  v8f acc = {};
  for (int k0 = 0; k0 < K; k0 += 32) {
    v16h a = ldfrag(A, K, arow, k0);
    v16h b = ldfrag(W, K, brow, k0);
    acc = WMMA_F16(a, b, acc);
  }
  int n = n0 + (lane & 15);
  float bv = bias ? bias[n] : 0.0f;
  int mb = m0 + 8 * (lane >> 4);
  for (int r = 0; r < 8; ++r) {
    float v = acc[r] + bv;
    if (do_tanh) v = tanhf(v);
    C[(size_t)(mb + r) * N + n] = v;
  }
}

// ---------------- recurrence: one persistent workgroup, 512 threads = 16 waves ----------------
__global__ void __launch_bounds__(512) rnnt_recurrence(
    const float* __restrict__ hs_pad, const int* __restrict__ hlens,
    const float* __restrict__ pre_enc, const _Float16* __restrict__ ey16,
    const _Float16* __restrict__ attdecW, const float* __restrict__ att_dec_b,
    const _Float16* __restrict__ Wih0, const _Float16* __restrict__ Whh0,
    const float* __restrict__ b_ih0, const float* __restrict__ b_hh0,
    const _Float16* __restrict__ Wih1, const _Float16* __restrict__ Whh1,
    const float* __restrict__ b_ih1, const float* __restrict__ b_hh1,
    _Float16* __restrict__ hdec16) {
  __shared__ alignas(16) _Float16 z0h[BB * DUNITS];   // 4KB
  __shared__ alignas(16) _Float16 z1h[BB * DUNITS];   // 4KB
  __shared__ alignas(16) _Float16 xh[BB * (EMB + EPROJS)]; // 6KB (concat [ey, att_c])
  __shared__ float c0s[BB * DUNITS];                  // 8KB
  __shared__ float c1s[BB * DUNITS];                  // 8KB
  __shared__ float dsh[BB * ATTD];                    // 8KB
  __shared__ float ew[BB * TT];                       // 3.2KB

  const int tid = threadIdx.x;
  const int lane = tid & 31;
  const int wave = tid >> 5;     // 0..15
  const int nlow = lane & 15;
  const int XK = EMB + EPROJS;   // 768

  for (int i = tid; i < BB * DUNITS; i += blockDim.x) {
    z0h[i] = (_Float16)0.0f; z1h[i] = (_Float16)0.0f;
    c0s[i] = 0.0f; c1s[i] = 0.0f;
  }
  __syncthreads();

  for (int u = 0; u < UU; ++u) {
    // --- d = tanh(z0 @ att_dec_W^T + b) ---
    for (int i = tid; i < BB * ATTD; i += blockDim.x) {
      int b = i >> 9, a = i & 511;
      const _Float16* wr = attdecW + (size_t)a * DUNITS;
      const _Float16* zr = z0h + b * DUNITS;
      float s = 0.0f;
      for (int h = 0; h < DUNITS; ++h) s += (float)zr[h] * (float)wr[h];
      dsh[i] = tanhf(s + att_dec_b[a]);
    }
    // copy embedding row into x
    for (int i = tid; i < BB * EMB; i += blockDim.x) {
      int b = i >> 8, k = i & (EMB - 1);
      xh[b * XK + k] = ey16[(size_t)(b * UU + u) * EMB + k];
    }
    __syncthreads();

    // --- e[b,t] = 2 * <pre_enc[b,t,:], d[b,:]>, masked ---
    for (int i = tid; i < BB * TT; i += blockDim.x) {
      int b = i / TT, t = i - b * TT;
      if (t < hlens[b]) {
        const float* pr = pre_enc + (size_t)(b * TT + t) * ATTD;
        const float* dr = dsh + b * ATTD;
        float s = 0.0f;
        for (int a = 0; a < ATTD; ++a) s += pr[a] * dr[a];
        ew[i] = 2.0f * s;
      } else {
        ew[i] = -1e30f;
      }
    }
    __syncthreads();

    // --- softmax over T, serialized per batch (tiny) ---
    if (tid < BB) {
      float m = -1e30f;
      for (int t = 0; t < TT; ++t) m = fmaxf(m, ew[tid * TT + t]);
      float s = 0.0f;
      for (int t = 0; t < TT; ++t) { float v = expf(ew[tid * TT + t] - m); ew[tid * TT + t] = v; s += v; }
      float inv = 1.0f / s;
      for (int t = 0; t < TT; ++t) ew[tid * TT + t] *= inv;
    }
    __syncthreads();

    // --- att_c[b,e] = sum_t w[b,t] * hs_pad[b,t,e]  (store as f16 into x) ---
    for (int i = tid; i < BB * EPROJS; i += blockDim.x) {
      int b = i >> 9, e = i & 511;
      float s = 0.0f;
      for (int t = 0; t < TT; ++t)
        s += ew[b * TT + t] * hs_pad[(size_t)(b * TT + t) * EPROJS + e];
      xh[b * XK + EMB + e] = (_Float16)s;
    }
    __syncthreads();

    // --- LSTM layer 0 gates via WMMA: each wave owns 2 chunks of 16 hidden units ---
    {
      v8f acc[2][4] = {};
      for (int k0 = 0; k0 < XK; k0 += 32) {
        v16h a = ldfrag(xh, XK, nlow & 3, k0);
        #pragma unroll
        for (int g = 0; g < 4; ++g)
          #pragma unroll
          for (int c = 0; c < 2; ++c) {
            v16h bg = ldfrag(Wih0, XK, g * DUNITS + wave * 16 + c * 256 + nlow, k0);
            acc[c][g] = WMMA_F16(a, bg, acc[c][g]);
          }
      }
      for (int k0 = 0; k0 < DUNITS; k0 += 32) {
        v16h a = ldfrag(z0h, DUNITS, nlow & 3, k0);
        #pragma unroll
        for (int g = 0; g < 4; ++g)
          #pragma unroll
          for (int c = 0; c < 2; ++c) {
            v16h bg = ldfrag(Whh0, DUNITS, g * DUNITS + wave * 16 + c * 256 + nlow, k0);
            acc[c][g] = WMMA_F16(a, bg, acc[c][g]);
          }
      }
      __syncthreads();   // all reads of old z0 done
      if (lane < 16) {
        #pragma unroll
        for (int c = 0; c < 2; ++c) {
          int unit = wave * 16 + c * 256 + lane;
          #pragma unroll
          for (int b = 0; b < 4; ++b) {
            float gi = acc[c][0][b] + b_ih0[unit] + b_hh0[unit];
            float gf = acc[c][1][b] + b_ih0[DUNITS + unit] + b_hh0[DUNITS + unit];
            float gg = acc[c][2][b] + b_ih0[2 * DUNITS + unit] + b_hh0[2 * DUNITS + unit];
            float go = acc[c][3][b] + b_ih0[3 * DUNITS + unit] + b_hh0[3 * DUNITS + unit];
            float cn = sigm(gf) * c0s[b * DUNITS + unit] + sigm(gi) * tanhf(gg);
            c0s[b * DUNITS + unit] = cn;
            z0h[b * DUNITS + unit] = (_Float16)(sigm(go) * tanhf(cn));
          }
        }
      }
      __syncthreads();
    }

    // --- LSTM layer 1 gates via WMMA (input = new z0, hidden = z1) ---
    {
      v8f acc[2][4] = {};
      for (int k0 = 0; k0 < DUNITS; k0 += 32) {
        v16h a = ldfrag(z0h, DUNITS, nlow & 3, k0);
        #pragma unroll
        for (int g = 0; g < 4; ++g)
          #pragma unroll
          for (int c = 0; c < 2; ++c) {
            v16h bg = ldfrag(Wih1, DUNITS, g * DUNITS + wave * 16 + c * 256 + nlow, k0);
            acc[c][g] = WMMA_F16(a, bg, acc[c][g]);
          }
      }
      for (int k0 = 0; k0 < DUNITS; k0 += 32) {
        v16h a = ldfrag(z1h, DUNITS, nlow & 3, k0);
        #pragma unroll
        for (int g = 0; g < 4; ++g)
          #pragma unroll
          for (int c = 0; c < 2; ++c) {
            v16h bg = ldfrag(Whh1, DUNITS, g * DUNITS + wave * 16 + c * 256 + nlow, k0);
            acc[c][g] = WMMA_F16(a, bg, acc[c][g]);
          }
      }
      __syncthreads();
      if (lane < 16) {
        #pragma unroll
        for (int c = 0; c < 2; ++c) {
          int unit = wave * 16 + c * 256 + lane;
          #pragma unroll
          for (int b = 0; b < 4; ++b) {
            float gi = acc[c][0][b] + b_ih1[unit] + b_hh1[unit];
            float gf = acc[c][1][b] + b_ih1[DUNITS + unit] + b_hh1[DUNITS + unit];
            float gg = acc[c][2][b] + b_ih1[2 * DUNITS + unit] + b_hh1[2 * DUNITS + unit];
            float go = acc[c][3][b] + b_ih1[3 * DUNITS + unit] + b_hh1[3 * DUNITS + unit];
            float cn = sigm(gf) * c1s[b * DUNITS + unit] + sigm(gi) * tanhf(gg);
            c1s[b * DUNITS + unit] = cn;
            _Float16 h16 = (_Float16)(sigm(go) * tanhf(cn));
            z1h[b * DUNITS + unit] = h16;
            hdec16[(size_t)(b * UU + u) * DUNITS + unit] = h16;
          }
        }
      }
      __syncthreads();
    }
  }
}

// ---------------- joint network: out[b,t,u,o] = tanh(henc+hdj) @ Wout^T + bout ----------------
// grid: (U/16, T, B), block: 256 threads (8 waves)
__global__ void __launch_bounds__(256) rnnt_joint(
    const float* __restrict__ henc, const float* __restrict__ hdj,
    const _Float16* __restrict__ WoutPad, const float* __restrict__ bout,
    float* __restrict__ out) {
  __shared__ alignas(16) _Float16 zt[16][JOINT + 8];  // stride 520 halves = 1040B (16B multiple)
  const int b = blockIdx.z, t = blockIdx.y, u0 = blockIdx.x * 16;
  const float* he = henc + (size_t)(b * TT + t) * JOINT;
  const float* hd = hdj + (size_t)(b * UU + u0) * JOINT;

  for (int idx = threadIdx.x; idx < 16 * JOINT; idx += blockDim.x) {
    int i = idx >> 9, j = idx & 511;
    zt[i][j] = (_Float16)tanhf(he[j] + hd[i * JOINT + j]);
  }
  __syncthreads();

  const int wave = threadIdx.x >> 5;
  const int lane = threadIdx.x & 31;
  const int arow = lane & 15;
  const int NOT = (ODIM_PAD / 16) - 1;  // 63 o-tiles cover 0..1007

  for (int ot = wave; ot < NOT; ot += 8) {
    int o0 = ot * 16;
    v8f acc = {};
    for (int k0 = 0; k0 < JOINT; k0 += 32) {
      v16h a = ldfrag(&zt[0][0], JOINT + 8, arow, k0);
      v16h w = ldfrag(WoutPad, JOINT, o0 + (lane & 15), k0);
      acc = WMMA_F16(a, w, acc);
    }
    int o = o0 + (lane & 15);
    if (o < ODIM) {
      float bv = bout[o];
      int ub = u0 + 8 * (lane >> 4);
      size_t base = (size_t)(b * TT + t) * UU;
      for (int r = 0; r < 8; ++r)
        out[(base + ub + r) * ODIM + o] = acc[r] + bv;
    }
  }
}

// ---------------- host ----------------
extern "C" void kernel_launch(void* const* d_in, const int* in_sizes, int n_in,
                              void* d_out, int out_size, void* d_ws, size_t ws_size,
                              hipStream_t stream) {
  const float* hs_pad    = (const float*)d_in[0];
  const int*   ys_in_pad = (const int*)d_in[1];
  const int*   hlens     = (const int*)d_in[2];
  const float* embed_W   = (const float*)d_in[3];
  const float* W_ih0     = (const float*)d_in[4];
  const float* W_hh0     = (const float*)d_in[5];
  const float* b_ih0     = (const float*)d_in[6];
  const float* b_hh0     = (const float*)d_in[7];
  const float* W_ih1     = (const float*)d_in[8];
  const float* W_hh1     = (const float*)d_in[9];
  const float* b_ih1     = (const float*)d_in[10];
  const float* b_hh1     = (const float*)d_in[11];
  const float* att_enc_W = (const float*)d_in[12];
  const float* att_enc_b = (const float*)d_in[13];
  const float* att_dec_W = (const float*)d_in[14];
  const float* att_dec_b = (const float*)d_in[15];
  const float* lin_enc_W = (const float*)d_in[16];
  const float* lin_enc_b = (const float*)d_in[17];
  const float* lin_dec_W = (const float*)d_in[18];
  const float* lin_out_W = (const float*)d_in[19];
  const float* lin_out_b = (const float*)d_in[20];
  float* out = (float*)d_out;

  size_t off = 0;
  auto alloc = [&](size_t bytes) -> void* {
    off = (off + 255) & ~(size_t)255;
    void* p = (char*)d_ws + off;
    off += bytes;
    return p;
  };
  _Float16* hs16       = (_Float16*)alloc((size_t)BB * TT * EPROJS * 2);
  _Float16* Wih0_16    = (_Float16*)alloc((size_t)4 * DUNITS * (EMB + EPROJS) * 2);
  _Float16* Whh0_16    = (_Float16*)alloc((size_t)4 * DUNITS * DUNITS * 2);
  _Float16* Wih1_16    = (_Float16*)alloc((size_t)4 * DUNITS * DUNITS * 2);
  _Float16* Whh1_16    = (_Float16*)alloc((size_t)4 * DUNITS * DUNITS * 2);
  _Float16* attencW16  = (_Float16*)alloc((size_t)ATTD * EPROJS * 2);
  _Float16* attdecW16  = (_Float16*)alloc((size_t)ATTD * DUNITS * 2);
  _Float16* linencW16  = (_Float16*)alloc((size_t)JOINT * EPROJS * 2);
  _Float16* lindecW16  = (_Float16*)alloc((size_t)JOINT * DUNITS * 2);
  _Float16* WoutPad16  = (_Float16*)alloc((size_t)ODIM_PAD * JOINT * 2);
  _Float16* ey16       = (_Float16*)alloc((size_t)BB * UU * EMB * 2);
  _Float16* hdec16     = (_Float16*)alloc((size_t)BB * UU * DUNITS * 2);
  float*    pre_enc    = (float*)alloc((size_t)BB * TT * ATTD * 4);
  float*    henc       = (float*)alloc((size_t)BB * TT * JOINT * 4);
  float*    hdj        = (float*)alloc((size_t)BB * UU * JOINT * 4);

  auto gblk = [](int n) { int g = (n + 255) / 256; return g > 4096 ? 4096 : g; };

  rnnt_cvt_f16<<<gblk(BB*TT*EPROJS), 256, 0, stream>>>(hs_pad, hs16, BB*TT*EPROJS);
  rnnt_cvt_f16<<<gblk(4*DUNITS*(EMB+EPROJS)), 256, 0, stream>>>(W_ih0, Wih0_16, 4*DUNITS*(EMB+EPROJS));
  rnnt_cvt_f16<<<gblk(4*DUNITS*DUNITS), 256, 0, stream>>>(W_hh0, Whh0_16, 4*DUNITS*DUNITS);
  rnnt_cvt_f16<<<gblk(4*DUNITS*DUNITS), 256, 0, stream>>>(W_ih1, Wih1_16, 4*DUNITS*DUNITS);
  rnnt_cvt_f16<<<gblk(4*DUNITS*DUNITS), 256, 0, stream>>>(W_hh1, Whh1_16, 4*DUNITS*DUNITS);
  rnnt_cvt_f16<<<gblk(ATTD*EPROJS), 256, 0, stream>>>(att_enc_W, attencW16, ATTD*EPROJS);
  rnnt_cvt_f16<<<gblk(ATTD*DUNITS), 256, 0, stream>>>(att_dec_W, attdecW16, ATTD*DUNITS);
  rnnt_cvt_f16<<<gblk(JOINT*EPROJS), 256, 0, stream>>>(lin_enc_W, linencW16, JOINT*EPROJS);
  rnnt_cvt_f16<<<gblk(JOINT*DUNITS), 256, 0, stream>>>(lin_dec_W, lindecW16, JOINT*DUNITS);
  rnnt_cvt_pad_rows<<<gblk(ODIM_PAD*JOINT), 256, 0, stream>>>(lin_out_W, WoutPad16, ODIM, ODIM_PAD, JOINT);
  rnnt_embed<<<gblk(BB*UU*EMB), 256, 0, stream>>>(embed_W, ys_in_pad, ey16);

  // pre_enc = tanh(hs @ att_enc_W^T + b): M=800, N=512, K=512
  rnnt_gemm_wmma<<<dim3(ATTD/16, (BB*TT)/16), 32, 0, stream>>>(
      hs16, attencW16, att_enc_b, pre_enc, EPROJS, ATTD, 1);
  // henc = hs @ lin_enc_W^T + b (no tanh)
  rnnt_gemm_wmma<<<dim3(JOINT/16, (BB*TT)/16), 32, 0, stream>>>(
      hs16, linencW16, lin_enc_b, henc, EPROJS, JOINT, 0);

  // sequential decoder recurrence (attention + 2 LSTM cells), 64 steps
  rnnt_recurrence<<<1, 512, 0, stream>>>(
      hs_pad, hlens, pre_enc, ey16, attdecW16, att_dec_b,
      Wih0_16, Whh0_16, b_ih0, b_hh0, Wih1_16, Whh1_16, b_ih1, b_hh1, hdec16);

  // hdj = hdec @ lin_dec_W^T (no bias): M=256, N=512, K=512
  rnnt_gemm_wmma<<<dim3(JOINT/16, (BB*UU)/16), 32, 0, stream>>>(
      hdec16, lindecW16, (const float*)nullptr, hdj, DUNITS, JOINT, 0);

  // joint: out = tanh(henc + hdj) @ lin_out_W^T + b
  rnnt_joint<<<dim3(UU/16, TT, BB), 256, 0, stream>>>(henc, hdj, WoutPad16, lin_out_b, out);

  (void)in_sizes; (void)n_in; (void)out_size; (void)ws_size;
}
